// Attention_1769526526676
// MI455X (gfx1250) — matrix operand order repaired and checked by reading, not compiled, not run on hardware
//
#include <hip/hip_runtime.h>

#define DEVINL __device__ __forceinline__

typedef __attribute__((ext_vector_type(16))) __bf16 bf16x16;
typedef __attribute__((ext_vector_type(8)))  __bf16 bf16x8;
typedef __attribute__((ext_vector_type(4)))  __bf16 bf16x4;
typedef __attribute__((ext_vector_type(8)))  float  fx8;

constexpr int Bn    = 32;
constexpr int Mn    = 512;
constexpr int En    = 256;
constexpr int Hn    = 8;
constexpr int An    = 64;
constexpr int ROWS  = Bn * Mn;   // 16384 rows of x
constexpr int NP    = Hn * An;   // 512 projection cols
constexpr int PAIRS = Hn * Bn;   // 256 (h,b) pairs
constexpr int CHUNK = Mn * An;   // 32768 elems per pair chunk

DEVINL bf16x16 make16(bf16x8 lo, bf16x8 hi) {
  return __builtin_shufflevector(lo, hi, 0, 1, 2, 3, 4, 5, 6, 7,
                                 8, 9, 10, 11, 12, 13, 14, 15);
}

// A fragment: 16(rows) x 32(K), row-major source with leading dim ld.
// ISA layout: lanes 0-15 row=lane K=0..7 & 16..23; lanes 16-31 K=8..15 & 24..31.
DEVINL bf16x16 load_frag_a(const __bf16* __restrict__ base, int row0, int ld,
                           int k0, int lane) {
  const __bf16* p =
      base + (size_t)(row0 + (lane & 15)) * ld + k0 + ((lane >> 4) << 3);
  bf16x8 lo = *(const bf16x8*)p;
  bf16x8 hi = *(const bf16x8*)(p + 16);
  return make16(lo, hi);
}

// B fragment: 32(K) x 16(cols), from B^T stored row-major [col][k], ld = K-dim.
// ISA layout: lanes 0-15 col=lane K=0..7 & 16..23; lanes 16-31 K=8..15 & 24..31.
DEVINL bf16x16 load_frag_b(const __bf16* __restrict__ baseT, int col0, int ld,
                           int k0, int lane) {
  const __bf16* p =
      baseT + (size_t)(col0 + (lane & 15)) * ld + k0 + ((lane >> 4) << 3);
  return make16(*(const bf16x8*)p, *(const bf16x8*)(p + 16));
}

DEVINL fx8 wmma_bf16(bf16x16 a, bf16x16 b, fx8 c) {
  return __builtin_amdgcn_wmma_f32_16x16x32_bf16(false, a, false, b, (short)0,
                                                 c, false, false);
}

// ---------------------------------------------------------------- f32 -> bf16
__global__ __launch_bounds__(256) void cvt_kernel(const float* __restrict__ in,
                                                  __bf16* __restrict__ out,
                                                  int n4) {
  int i = blockIdx.x * blockDim.x + threadIdx.x;
  if (i < n4) {
    float4 v = ((const float4*)in)[i];
    bf16x4 o = {(__bf16)v.x, (__bf16)v.y, (__bf16)v.z, (__bf16)v.w};
    *(bf16x4*)(out + (size_t)i * 4) = o;
  }
}

// --------------------------------------- W (RxC f32) -> W^T (CxR bf16), tiled
__global__ __launch_bounds__(256) void transcvt_kernel(
    const float* __restrict__ in, __bf16* __restrict__ out, int R, int C) {
  __shared__ float t[64][65];
  const int r0 = blockIdx.x * 64, c0 = blockIdx.y * 64;
  const int tr = threadIdx.x >> 6;   // 0..3
  const int tc = threadIdx.x & 63;
#pragma unroll
  for (int j = 0; j < 16; ++j) {
    int r = tr + j * 4;
    t[r][tc] = in[(size_t)(r0 + r) * C + c0 + tc];
  }
  __syncthreads();
#pragma unroll
  for (int j = 0; j < 16; ++j) {
    int r = tr + j * 4;
    out[(size_t)(c0 + r) * R + r0 + tc] = (__bf16)t[tc][r];
  }
}

// ----------------------- per-pair bf16 transpose: chunk (RxC) -> chunk (CxR)
__global__ __launch_bounds__(256) void transpose_pair(
    const __bf16* __restrict__ in, __bf16* __restrict__ out, int R, int C) {
  __shared__ __bf16 t[64][72];
  const int p = blockIdx.x;
  const int r0 = blockIdx.y * 64, c0 = blockIdx.z * 64;
  const __bf16* ip = in + (size_t)p * R * C;
  __bf16* op = out + (size_t)p * R * C;
  const int tr = threadIdx.x >> 6;
  const int tc = threadIdx.x & 63;
#pragma unroll
  for (int j = 0; j < 16; ++j) {
    int r = tr + j * 4;
    t[r][tc] = ip[(size_t)(r0 + r) * C + c0 + tc];
  }
  __syncthreads();
#pragma unroll
  for (int j = 0; j < 16; ++j) {
    int r = tr + j * 4;
    op[(size_t)(c0 + r) * R + r0 + tc] = t[tc][r];
  }
}

// --------------------------- projection GEMM: C(ROWSxN) = X(ROWSxE) @ W(ExN)
// WG = 256 thr (8 waves), waves tiled 2(m) x 4(n): WG covers 128m x 64n.
template <bool OUT_BF16>
__global__ __launch_bounds__(256) void proj_kernel(
    const __bf16* __restrict__ X, const __bf16* __restrict__ WT,
    __bf16* __restrict__ outB, float* __restrict__ outF, int N) {
  const int lane = threadIdx.x & 31;
  const int w = threadIdx.x >> 5;
  const int m0 = blockIdx.x * 128 + (w & 1) * 64;
  const int c0 = blockIdx.y * 64 + (w >> 1) * 16;

  fx8 zero = {};
  fx8 acc[4] = {zero, zero, zero, zero};
#pragma unroll
  for (int k0 = 0; k0 < En; k0 += 32) {
    bf16x16 bf = load_frag_b(WT, c0, En, k0, lane);
#pragma unroll
    for (int mt = 0; mt < 4; ++mt) {
      bf16x16 af = load_frag_a(X, m0 + mt * 16, En, k0, lane);
      acc[mt] = wmma_bf16(af, bf, acc[mt]);
    }
  }
  // D layout: VGPR i -> (row i | i+8 per lane half), col = lane&15
#pragma unroll
  for (int mt = 0; mt < 4; ++mt) {
#pragma unroll
    for (int i = 0; i < 8; ++i) {
      const int row = m0 + mt * 16 + ((lane >> 4) << 3) + i;
      const int col = c0 + (lane & 15);
      const size_t idx = (size_t)row * N + col;
      if (OUT_BF16)
        outB[idx] = (__bf16)acc[mt][i];
      else
        outF[idx] = acc[mt][i];
    }
  }
}

// ----------------------------------------------------------------- attention
// One WG per (b, 32-row m-block). Loops h: head-mean accumulates in C operand.
__global__ __launch_bounds__(256) void attn_kernel(
    const __bf16* __restrict__ Q, const __bf16* __restrict__ Kt,
    const __bf16* __restrict__ Vt, const float* __restrict__ Rf,
    float* __restrict__ out) {
  __shared__ __bf16 att[32][520];   // normalized probabilities, padded rows
  __shared__ float red[8 * 32];     // per-wave partial max/sum
  __shared__ float rowstat[32];     // final row max, then row sum

  const int lane = threadIdx.x & 31;
  const int w = threadIdx.x >> 5;
  const int half = lane >> 4;
  const int b = blockIdx.x >> 4;
  const int m0 = (blockIdx.x & 15) * 32;

  const int n0 = w * 64;       // this wave's 64-col slab of S
  const int mt2 = w >> 2;      // GEMM2 out-tile row (0..1)
  const int at = w & 3;        // GEMM2 out-tile col (0..3)

  fx8 zero = {};
  fx8 oacc = zero;

  for (int h = 0; h < Hn; ++h) {
    const size_t poff = (size_t)(h * Bn + b) * CHUNK;
    const __bf16* qb = Q + poff;   // (512 m, 64 a)
    const __bf16* kb = Kt + poff;  // (512 n, 64 a)  == K^T per pair
    const __bf16* vb = Vt + poff;  // (64 a, 512 n)  == V^T per pair

    // ---- S slab (32m x 64n), K-loop over a = 0..63
    fx8 s[2][4];
#pragma unroll
    for (int mt = 0; mt < 2; ++mt)
#pragma unroll
      for (int nt = 0; nt < 4; ++nt) s[mt][nt] = zero;

#pragma unroll
    for (int a0 = 0; a0 < An; a0 += 32) {
      bf16x16 bk[4];
#pragma unroll
      for (int nt = 0; nt < 4; ++nt)
        bk[nt] = load_frag_b(kb, n0 + nt * 16, An, a0, lane);
#pragma unroll
      for (int mt = 0; mt < 2; ++mt) {
        bf16x16 af = load_frag_a(qb, m0 + mt * 16, An, a0, lane);
#pragma unroll
        for (int nt = 0; nt < 4; ++nt)
          s[mt][nt] = wmma_bf16(af, bk[nt], s[mt][nt]);
      }
    }

    __syncthreads();  // previous head finished reading att/red

    // ---- row max: intra-wave (4 tiles + 16-lane shuffle), then cross-wave
#pragma unroll
    for (int mt = 0; mt < 2; ++mt) {
#pragma unroll
      for (int i = 0; i < 8; ++i) {
        float v = fmaxf(fmaxf(s[mt][0][i], s[mt][1][i]),
                        fmaxf(s[mt][2][i], s[mt][3][i]));
#pragma unroll
        for (int off = 8; off >= 1; off >>= 1)
          v = fmaxf(v, __shfl_xor(v, off, 16));
        if ((lane & 15) == 0) red[w * 32 + mt * 16 + half * 8 + i] = v;
      }
    }
    __syncthreads();
    if (threadIdx.x < 32) {
      float v = red[threadIdx.x];
#pragma unroll
      for (int ww = 1; ww < 8; ++ww)
        v = fmaxf(v, red[ww * 32 + threadIdx.x]);
      rowstat[threadIdx.x] = v;
    }
    __syncthreads();

    // ---- exp + row sum
#pragma unroll
    for (int mt = 0; mt < 2; ++mt) {
#pragma unroll
      for (int i = 0; i < 8; ++i) {
        const int row = mt * 16 + half * 8 + i;
        const float rm = rowstat[row];
        float acc = 0.f;
#pragma unroll
        for (int nt = 0; nt < 4; ++nt) {
          float e = __expf(s[mt][nt][i] - rm);
          s[mt][nt][i] = e;
          acc += e;
        }
#pragma unroll
        for (int off = 8; off >= 1; off >>= 1)
          acc += __shfl_xor(acc, off, 16);
        if ((lane & 15) == 0) red[w * 32 + row] = acc;
      }
    }
    __syncthreads();
    if (threadIdx.x < 32) {
      float v = 0.f;
#pragma unroll
      for (int ww = 0; ww < 8; ++ww) v += red[ww * 32 + threadIdx.x];
      rowstat[threadIdx.x] = v;
    }
    __syncthreads();

    // ---- normalize, store bf16 probabilities to LDS
#pragma unroll
    for (int mt = 0; mt < 2; ++mt) {
#pragma unroll
      for (int i = 0; i < 8; ++i) {
        const int row = mt * 16 + half * 8 + i;
        const float inv = __builtin_amdgcn_rcpf(rowstat[row]);
#pragma unroll
        for (int nt = 0; nt < 4; ++nt)
          att[row][n0 + nt * 16 + (lane & 15)] = (__bf16)(s[mt][nt][i] * inv);
      }
    }
    __syncthreads();

    // ---- out += att(32x512) @ V(512x64); one 16x16 tile per wave
    for (int k0 = 0; k0 < Mn; k0 += 32) {
      const __bf16* pa = &att[mt2 * 16 + (lane & 15)][k0 + (half << 3)];
      bf16x8 lo = *(const bf16x8*)pa;
      bf16x8 hi = *(const bf16x8*)(pa + 16);
      bf16x16 af = make16(lo, hi);
      if (k0 + 32 < Mn)
        __builtin_prefetch(vb + (size_t)(at * 16 + (lane & 15)) * Mn + k0 + 32,
                           0, 1);
      bf16x16 bf = load_frag_b(vb, at * 16, Mn, k0, lane);
      oacc = wmma_bf16(af, bf, oacc);
    }
  }

  // ---- epilogue: head mean + x@Wr + ReLU
#pragma unroll
  for (int i = 0; i < 8; ++i) {
    const int ml = mt2 * 16 + half * 8 + i;
    const size_t r =
        ((size_t)b * Mn + m0 + ml) * An + at * 16 + (lane & 15);
    float v = oacc[i] * 0.125f + Rf[r];
    out[r] = v > 0.f ? v : 0.f;
  }
}

// ------------------------------------------------------------------- launch
extern "C" void kernel_launch(void* const* d_in, const int* in_sizes, int n_in,
                              void* d_out, int out_size, void* d_ws,
                              size_t ws_size, hipStream_t stream) {
  (void)in_sizes; (void)n_in; (void)out_size; (void)ws_size;

  const float* x  = (const float*)d_in[0];
  const float* Wq = (const float*)d_in[1];
  const float* Wk = (const float*)d_in[2];
  const float* Wv = (const float*)d_in[3];
  const float* Wr = (const float*)d_in[4];
  float* out = (float*)d_out;

  char* ws = (char*)d_ws;
  size_t off = 0;
  auto take = [&](size_t bytes) -> char* {
    char* p = ws + off;
    off += (bytes + 255) & ~(size_t)255;
    return p;
  };

  __bf16* xb  = (__bf16*)take((size_t)ROWS * En * 2);  // x in bf16
  __bf16* WqT = (__bf16*)take((size_t)NP * En * 2);    // W^T bf16
  __bf16* WkT = (__bf16*)take((size_t)NP * En * 2);
  __bf16* WvT = (__bf16*)take((size_t)NP * En * 2);
  __bf16* WrT = (__bf16*)take((size_t)An * En * 2);
  __bf16* Qb  = (__bf16*)take((size_t)ROWS * NP * 2);  // Q natural
  __bf16* Kn  = (__bf16*)take((size_t)ROWS * NP * 2);  // K natural
  __bf16* Ktb = (__bf16*)take((size_t)ROWS * NP * 2);  // K^T per pair
  __bf16* Vn  = (__bf16*)take((size_t)ROWS * NP * 2);  // V natural
  __bf16* Vtb = (__bf16*)take((size_t)ROWS * NP * 2);  // V^T per pair
  float*  Rf  = (float*)take((size_t)ROWS * An * 4);   // x @ Wr (f32)

  cvt_kernel<<<(ROWS * En / 4) / 256, 256, 0, stream>>>(x, xb, ROWS * En / 4);

  transcvt_kernel<<<dim3(En / 64, NP / 64), 256, 0, stream>>>(Wq, WqT, En, NP);
  transcvt_kernel<<<dim3(En / 64, NP / 64), 256, 0, stream>>>(Wk, WkT, En, NP);
  transcvt_kernel<<<dim3(En / 64, NP / 64), 256, 0, stream>>>(Wv, WvT, En, NP);
  transcvt_kernel<<<dim3(En / 64, An / 64), 256, 0, stream>>>(Wr, WrT, En, An);

  proj_kernel<true><<<dim3(ROWS / 128, NP / 64), 256, 0, stream>>>(
      xb, WqT, Qb, nullptr, NP);
  proj_kernel<true><<<dim3(ROWS / 128, NP / 64), 256, 0, stream>>>(
      xb, WkT, Kn, nullptr, NP);
  proj_kernel<true><<<dim3(ROWS / 128, NP / 64), 256, 0, stream>>>(
      xb, WvT, Vn, nullptr, NP);
  proj_kernel<false><<<dim3(ROWS / 128, An / 64), 256, 0, stream>>>(
      xb, WrT, nullptr, Rf, An);

  // K pair chunk (64 x 512) -> (512 x 64); V pair chunk (512 x 64) -> (64 x 512)
  transpose_pair<<<dim3(PAIRS, An / 64, Mn / 64), 256, 0, stream>>>(Kn, Ktb, An, Mn);
  transpose_pair<<<dim3(PAIRS, Mn / 64, An / 64), 256, 0, stream>>>(Vn, Vtb, Mn, An);

  attn_kernel<<<Bn * (Mn / 32), 256, 0, stream>>>(Qb, Ktb, Vtb, Rf, out);
}
